// AttnDecoderSRU_9225589752589
// MI455X (gfx1250) — matrix operand order repaired and checked by reading, not compiled, not run on hardware
//
#include <hip/hip_runtime.h>
#include <hip/hip_bf16.h>
#include <stdint.h>

// ---------------------------------------------------------------------------
// AttnDecoderSRU on gfx1250 (MI455X).
// bf16 operands + f32 accumulate via v_wmma_f32_16x16x32_bf16.
// GEMM: 256-thread blocks, 8 waves, wave tile 32x64, double-buffered LDS,
// async global->LDS staging (ASYNCcnt) overlapping the WMMA stream.
// ---------------------------------------------------------------------------

#define L_ 64
#define B_ 32
#define S_ 64
#define D_ 512
#define V_ 32000
#define NL_ 2

#define USE_ASYNC_LDS 1

typedef __attribute__((ext_vector_type(16))) __bf16 v16bf;
typedef __attribute__((ext_vector_type(8)))  float  v8f;

__device__ __forceinline__ unsigned short f2bf(float f) {
  union { float f; uint32_t u; } c; c.f = f;
  uint32_t u = c.u;
  uint32_t r = u + 0x7FFFu + ((u >> 16) & 1u);   // round-to-nearest-even
  return (unsigned short)(r >> 16);
}

__device__ __forceinline__ float sigmoidf_(float x) {
  return 1.0f / (1.0f + __expf(-x));
}

__device__ __forceinline__ void async_ld_b128(unsigned lds_addr, const void* gaddr) {
#if USE_ASYNC_LDS
  asm volatile("global_load_async_to_lds_b128 %0, %1, off"
               :: "v"(lds_addr), "v"((unsigned long long)(uintptr_t)gaddr)
               : "memory");
#endif
}

__device__ __forceinline__ void wait_async0() {
#if USE_ASYNC_LDS
  asm volatile("s_wait_asynccnt 0" ::: "memory");
#endif
}

// ---------------------------------------------------------------------------
// C[M,N] (f32 and/or bf16) = act( A[M,K](bf16,row-major) * Bt[N,K](bf16) + bias )
// Block tile 256(M) x 64(N); wave tile 32 x 64 (8 accumulators).
// B tile (64x32 bf16 = 4KB) double-buffered in LDS; staged with async DMA.
// Batched via blockIdx.z with element strides.
// ---------------------------------------------------------------------------
__global__ __launch_bounds__(256) void gemm_bf16_wmma(
    const unsigned short* __restrict__ A,
    const unsigned short* __restrict__ Bt,
    float* __restrict__ Cf,
    unsigned short* __restrict__ Cb,
    const float* __restrict__ bias,
    int M, int N, int K,
    long strideA, long strideB, long strideC,
    int act_tanh)
{
  __shared__ __align__(16) unsigned short bs[2][64 * 32];  // 2 x 4KB

  const int bz = blockIdx.z;
  A  += (long)bz * strideA;
  Bt += (long)bz * strideB;

  const int tid  = threadIdx.x;
  const int lane = tid & 31;
  const int wave = tid >> 5;
  const int half = (lane >> 4) & 1;    // 0 = lanes 0-15, 1 = lanes 16-31
  const int l16  = lane & 15;

  const int n0 = blockIdx.x * 64;
  const int m0 = blockIdx.y * 256 + wave * 32;

  // two A rows per lane (wave covers 32 M rows)
  int ar0 = m0 + l16;        if (ar0 >= M) ar0 = 0;   // clamp; stores masked
  int ar1 = m0 + 16 + l16;   if (ar1 >= M) ar1 = 0;
  const unsigned short* aptr0 = A + (long)ar0 * K + half * 8;
  const unsigned short* aptr1 = A + (long)ar1 * K + half * 8;

  // staging: thread t moves 16B of Bt row (n0 + t/4), K-segment (t&3)*8
  const int srow = tid >> 2;           // 0..63
  const int sseg = tid & 3;            // 0..3
  const unsigned short* bload = Bt + (long)(n0 + srow) * K + sseg * 8;
  const unsigned lds_slot = (unsigned)(srow * 32 + sseg * 8);
  const unsigned lds_base0 = (unsigned)(uintptr_t)&bs[0][lds_slot];
  const unsigned lds_base1 = (unsigned)(uintptr_t)&bs[1][lds_slot];

  v8f acc[2][4];
#pragma unroll
  for (int mi = 0; mi < 2; ++mi)
#pragma unroll
    for (int g = 0; g < 4; ++g) acc[mi][g] = v8f{};

  // preload tile k=0 into buffer 0
#if USE_ASYNC_LDS
  async_ld_b128(lds_base0, bload);
  wait_async0();
#else
  *(uint4*)&bs[0][lds_slot] = *(const uint4*)bload;
#endif
  __syncthreads();

  int cur = 0;
  for (int k = 0; k < K; k += 32) {
    const int nxt = cur ^ 1;
    const bool more = (k + 32) < K;

#if USE_ASYNC_LDS
    if (more) async_ld_b128(nxt ? lds_base1 : lds_base0, bload + (k + 32));
#else
    uint4 breg;
    if (more) breg = *(const uint4*)(bload + (k + 32));
#endif

    // A fragments (global, b128 x4)
    v16bf a0, a1;
    {
      uint4* p = (uint4*)&a0;
      p[0] = *(const uint4*)(aptr0 + k);
      p[1] = *(const uint4*)(aptr0 + k + 16);
      uint4* q = (uint4*)&a1;
      q[0] = *(const uint4*)(aptr1 + k);
      q[1] = *(const uint4*)(aptr1 + k + 16);
    }

    // B fragments from LDS (shared by both M-subtiles)
    v16bf bfrag[4];
#pragma unroll
    for (int g = 0; g < 4; ++g) {
      const unsigned short* p = &bs[cur][(g * 16 + l16) * 32 + half * 8];
      uint4* q = (uint4*)&bfrag[g];
      q[0] = ((const uint4*)p)[0];
      q[1] = ((const uint4*)p)[2];
    }

#pragma unroll
    for (int g = 0; g < 4; ++g) {
      acc[0][g] = __builtin_amdgcn_wmma_f32_16x16x32_bf16(
          false, a0, false, bfrag[g], (short)0, acc[0][g], false, false);
      acc[1][g] = __builtin_amdgcn_wmma_f32_16x16x32_bf16(
          false, a1, false, bfrag[g], (short)0, acc[1][g], false, false);
    }

#if USE_ASYNC_LDS
    if (more) wait_async0();
#else
    if (more) *(uint4*)&bs[nxt][lds_slot] = breg;
#endif
    __syncthreads();
    cur = nxt;
  }

  // C layout: VGPR i -> M = m0 + mi*16 + half*8 + i ; N = n0 + g*16 + l16
#pragma unroll
  for (int mi = 0; mi < 2; ++mi) {
#pragma unroll
    for (int i = 0; i < 8; ++i) {
      const int m = m0 + mi * 16 + half * 8 + i;
      if (m >= M) continue;
#pragma unroll
      for (int g = 0; g < 4; ++g) {
        const int n = n0 + g * 16 + l16;
        float v = acc[mi][g][i];
        if (bias) v += bias[n];
        if (act_tanh) v = tanhf(v);
        const long idx = (long)bz * strideC + (long)m * N + n;
        if (Cf) Cf[idx] = v;
        if (Cb) Cb[idx] = f2bf(v);
      }
    }
  }
}

// ---------------------------------------------------------------------------
// Embedding gather: x[L*B, D] (f32 + bf16) from table[V, D] and idx[L*B]
// ---------------------------------------------------------------------------
__global__ void embed_kernel(const int* __restrict__ idx,
                             const float* __restrict__ table,
                             float* __restrict__ Xf,
                             unsigned short* __restrict__ Xb)
{
  const int i = blockIdx.x * blockDim.x + threadIdx.x;
  if (i >= L_ * B_ * D_) return;
  const int d  = i & (D_ - 1);
  const int lb = i >> 9;                 // D_ == 512
  const float v = table[(long)idx[lb] * D_ + d];
  Xf[i] = v;
  Xb[i] = f2bf(v);
}

// W[K,N] f32 -> Wt[N,K] bf16
__global__ void transpose_bf16_kernel(const float* __restrict__ W,
                                      unsigned short* __restrict__ Wt,
                                      int K, int N)
{
  const long i = (long)blockIdx.x * blockDim.x + threadIdx.x;
  if (i >= (long)K * N) return;
  const int k = (int)(i / N);
  const int n = (int)(i % N);
  Wt[(long)n * K + k] = f2bf(W[i]);
}

// encoder_outputs[S,B,D] -> memA[B,S,D] bf16 (scores) and memB[B,D,S] bf16 (ctx)
__global__ void enc_transpose_kernel(const float* __restrict__ E,
                                     unsigned short* __restrict__ memA,
                                     unsigned short* __restrict__ memB)
{
  const long i = (long)blockIdx.x * blockDim.x + threadIdx.x;
  if (i >= (long)S_ * B_ * D_) return;
  const int d = (int)(i % D_);
  const long r = i / D_;
  const int b = (int)(r % B_);
  const int s = (int)(r / B_);
  const unsigned short v = f2bf(E[i]);
  memA[((long)b * S_ + s) * D_ + d] = v;
  memB[((long)b * D_ + d) * S_ + s] = v;
}

// ---------------------------------------------------------------------------
// SRU recurrence: one thread per (b,d), serial over L.
// ---------------------------------------------------------------------------
__global__ __launch_bounds__(256) void sru_scan_kernel(
    const float* __restrict__ U,     // [L*B, 3D]
    const float* __restrict__ X,     // [L*B, D]
    const float* __restrict__ c0,    // [B, D]
    const float* __restrict__ bvec,  // [2D]
    float* __restrict__ H,           // [L*B, D]
    unsigned short* __restrict__ Hb, // [L*B, D] bf16
    float* __restrict__ c_out)       // [B, D]
{
  const int i = blockIdx.x * 256 + threadIdx.x;   // 0 .. B*D-1
  if (i >= B_ * D_) return;
  const int b = i >> 9;
  const int d = i & (D_ - 1);
  float c = c0[i];
  const float bf_ = bvec[d];
  const float br_ = bvec[D_ + d];
  for (int l = 0; l < L_; ++l) {
    const long row = (long)(l * B_ + b);
    const float* u = U + row * (3 * D_);
    const float z = u[d];
    const float f = sigmoidf_(u[D_ + d] + bf_);
    const float r = sigmoidf_(u[2 * D_ + d] + br_);
    c = f * c + (1.0f - f) * z;
    const float x = X[row * D_ + d];
    const float h = r * tanhf(c) + (1.0f - r) * x;
    H[row * D_ + d]  = h;
    Hb[row * D_ + d] = f2bf(h);
  }
  c_out[i] = c;
}

// Hb[L,B,D] bf16 -> Q[B,L,D] bf16
__global__ void reorder_q_kernel(const unsigned short* __restrict__ Hb,
                                 unsigned short* __restrict__ Q)
{
  const int i = blockIdx.x * blockDim.x + threadIdx.x;
  if (i >= L_ * B_ * D_) return;
  const int d  = i & (D_ - 1);
  const int lb = i >> 9;
  const int b  = lb & (B_ - 1);
  const int l  = lb >> 5;                // B_ == 32
  Q[((long)(b * L_ + l)) * D_ + d] = Hb[i];
}

// Row softmax over S=64: one wave per row; outputs bf16 weights.
__global__ __launch_bounds__(256) void softmax64_kernel(
    const float* __restrict__ SC, unsigned short* __restrict__ AL, int rows)
{
  const int row  = blockIdx.x * 8 + (threadIdx.x >> 5);
  const int lane = threadIdx.x & 31;
  if (row >= rows) return;
  const float* p = SC + (long)row * S_;
  float v0 = p[lane], v1 = p[lane + 32];
  float m = fmaxf(v0, v1);
#pragma unroll
  for (int off = 16; off > 0; off >>= 1) m = fmaxf(m, __shfl_xor(m, off, 32));
  const float e0 = __expf(v0 - m);
  const float e1 = __expf(v1 - m);
  float s = e0 + e1;
#pragma unroll
  for (int off = 16; off > 0; off >>= 1) s += __shfl_xor(s, off, 32);
  const float inv = 1.0f / s;
  unsigned short* o = AL + (long)row * S_;
  o[lane]      = f2bf(e0 * inv);
  o[lane + 32] = f2bf(e1 * inv);
}

// cat[B*L, 2D] bf16 = [ ctx[B*L, D] f32 | q from H2[L,B,D] f32 ]
__global__ void concat_kernel(const float* __restrict__ CTX,
                              const float* __restrict__ H2,
                              unsigned short* __restrict__ CAT)
{
  const long i = (long)blockIdx.x * blockDim.x + threadIdx.x;
  if (i >= (long)B_ * L_ * 2 * D_) return;
  const int d2 = (int)(i & (2 * D_ - 1));
  const long bl = i >> 10;               // 2D == 1024
  float v;
  if (d2 < D_) {
    v = CTX[bl * D_ + d2];
  } else {
    const int b = (int)(bl / L_);
    const int l = (int)(bl % L_);
    v = H2[((long)(l * B_ + b)) * D_ + (d2 - D_)];
  }
  CAT[i] = f2bf(v);
}

// ---------------------------------------------------------------------------
// Host-side orchestration
// ---------------------------------------------------------------------------
extern "C" void kernel_launch(void* const* d_in, const int* in_sizes, int n_in,
                              void* d_out, int out_size, void* d_ws, size_t ws_size,
                              hipStream_t stream) {
  const int*   rnn_input   = (const int*)  d_in[0];   // [L,B]
  const float* last_hidden = (const float*)d_in[1];   // [NL,B,D]
  const float* enc_out     = (const float*)d_in[2];   // [S,B,D]
  const float* embed_table = (const float*)d_in[3];   // [V,D]
  const float* sru_W       = (const float*)d_in[4];   // [NL,D,3D]
  const float* sru_b       = (const float*)d_in[5];   // [NL,2D]
  const float* attn_W      = (const float*)d_in[6];   // [2D,D]
  const float* out_W       = (const float*)d_in[7];   // [D,V]
  const float* out_b       = (const float*)d_in[8];   // [V]

  float* out_logits = (float*)d_out;                       // [B,L,V]
  float* out_hidden = (float*)d_out + (long)B_ * L_ * V_;  // [NL,B,D]

  char* w = (char*)d_ws;
  size_t off = 0;
  auto take = [&](size_t bytes) -> void* {
    void* p = w + off;
    off += (bytes + 255) & ~(size_t)255;
    return p;
  };
  float*          Xf0  = (float*)take((size_t)L_ * B_ * D_ * 4);
  float*          Xf1  = (float*)take((size_t)L_ * B_ * D_ * 4);
  unsigned short* Xb0  = (unsigned short*)take((size_t)L_ * B_ * D_ * 2);
  unsigned short* Xb1  = (unsigned short*)take((size_t)L_ * B_ * D_ * 2);
  float*          U    = (float*)take((size_t)L_ * B_ * 3 * D_ * 4);
  unsigned short* WtS0 = (unsigned short*)take((size_t)3 * D_ * D_ * 2);
  unsigned short* WtS1 = (unsigned short*)take((size_t)3 * D_ * D_ * 2);
  unsigned short* WtA  = (unsigned short*)take((size_t)D_ * 2 * D_ * 2);
  unsigned short* WtO  = (unsigned short*)take((size_t)V_ * D_ * 2);
  unsigned short* memA = (unsigned short*)take((size_t)B_ * S_ * D_ * 2);
  unsigned short* memB = (unsigned short*)take((size_t)B_ * D_ * S_ * 2);
  unsigned short* Q    = (unsigned short*)take((size_t)B_ * L_ * D_ * 2);
  float*          SCO  = (float*)take((size_t)B_ * L_ * S_ * 4);
  unsigned short* ALI  = (unsigned short*)take((size_t)B_ * L_ * S_ * 2);
  float*          CTX  = (float*)take((size_t)B_ * L_ * D_ * 4);
  unsigned short* CAT  = (unsigned short*)take((size_t)B_ * L_ * 2 * D_ * 2);
  unsigned short* AHb  = (unsigned short*)take((size_t)B_ * L_ * D_ * 2);
  (void)ws_size; (void)n_in; (void)in_sizes; (void)out_size;

  const int T = 256;

  // 1. embedding gather (f32 + bf16)
  embed_kernel<<<(L_ * B_ * D_ + T - 1) / T, T, 0, stream>>>(rnn_input, embed_table, Xf0, Xb0);

  // 2. weights: f32 [K,N] -> bf16 [N,K]
  transpose_bf16_kernel<<<(D_ * 3 * D_ + T - 1) / T, T, 0, stream>>>(sru_W,             WtS0, D_,     3 * D_);
  transpose_bf16_kernel<<<(D_ * 3 * D_ + T - 1) / T, T, 0, stream>>>(sru_W + (long)D_ * 3 * D_, WtS1, D_, 3 * D_);
  transpose_bf16_kernel<<<(2 * D_ * D_ + T - 1) / T, T, 0, stream>>>(attn_W,            WtA,  2 * D_, D_);
  transpose_bf16_kernel<<<((long)D_ * V_ + T - 1) / T, T, 0, stream>>>(out_W,           WtO,  D_,     V_);

  // 3. encoder memory layouts
  enc_transpose_kernel<<<(S_ * B_ * D_ + T - 1) / T, T, 0, stream>>>(enc_out, memA, memB);

  // 4. SRU layers
  float*          Xf[2] = {Xf0, Xf1};
  unsigned short* Xb[2] = {Xb0, Xb1};
  unsigned short* WtS[2] = {WtS0, WtS1};
  int cur = 0;
  for (int l = 0; l < NL_; ++l) {
    const int nxt = cur ^ 1;
    // U = x @ W  : M=2048, N=1536, K=512
    gemm_bf16_wmma<<<dim3(3 * D_ / 64, (L_ * B_) / 256, 1), T, 0, stream>>>(
        Xb[cur], WtS[l], U, nullptr, nullptr,
        L_ * B_, 3 * D_, D_, 0, 0, 0, 0);
    sru_scan_kernel<<<(B_ * D_ + T - 1) / T, T, 0, stream>>>(
        U, Xf[cur], last_hidden + (long)l * B_ * D_, sru_b + (long)l * 2 * D_,
        Xf[nxt], Xb[nxt], out_hidden + (long)l * B_ * D_);
    cur = nxt;
  }

  // 5. q in (b,l) row order
  reorder_q_kernel<<<(L_ * B_ * D_ + T - 1) / T, T, 0, stream>>>(Xb[cur], Q);

  // 6. scores[b] = q[b] @ mem[b]^T : batched M=64, N=64, K=512
  gemm_bf16_wmma<<<dim3(1, 1, B_), T, 0, stream>>>(
      Q, memA, SCO, nullptr, nullptr,
      L_, S_, D_, (long)L_ * D_, (long)S_ * D_, (long)L_ * S_, 0);

  // 7. softmax over S
  softmax64_kernel<<<(B_ * L_ + 7) / 8, T, 0, stream>>>(SCO, ALI, B_ * L_);

  // 8. ctx[b] = align[b] @ mem[b] : batched M=64, N=512, K=64
  gemm_bf16_wmma<<<dim3(D_ / 64, 1, B_), T, 0, stream>>>(
      ALI, memB, CTX, nullptr, nullptr,
      L_, D_, S_, (long)L_ * S_, (long)D_ * S_, (long)L_ * D_, 0);

  // 9. cat = [ctx | q]
  concat_kernel<<<((long)B_ * L_ * 2 * D_ + T - 1) / T, T, 0, stream>>>(CTX, Xf[cur], CAT);

  // 10. attn_h = tanh(cat @ attn_W) : M=2048, N=512, K=1024 (bf16 out)
  gemm_bf16_wmma<<<dim3(D_ / 64, (B_ * L_) / 256, 1), T, 0, stream>>>(
      CAT, WtA, nullptr, AHb, nullptr,
      B_ * L_, D_, 2 * D_, 0, 0, 0, 1);

  // 11. logits = attn_h @ out_W + out_b : M=2048, N=32000, K=512 (f32 to d_out)
  gemm_bf16_wmma<<<dim3(V_ / 64, (B_ * L_) / 256, 1), T, 0, stream>>>(
      AHb, WtO, out_logits, nullptr, out_b,
      B_ * L_, V_, D_, 0, 0, 0, 0);
}
